// SiglipPoolingHead_PatchCLS_77369540870731
// MI455X (gfx1250) — matrix operand order repaired and checked by reading, not compiled, not run on hardware
//
#include <hip/hip_runtime.h>
#include <math.h>

#define DEV static __device__ __forceinline__

typedef __bf16 bf16_t;
typedef __attribute__((ext_vector_type(16))) __bf16 v16bf;
typedef __attribute__((ext_vector_type(8)))  float  v8f;
typedef int v4i __attribute__((vector_size(16)));   // matches async builtin prototype

// Problem dims
constexpr int B_ = 16, S_ = 729, D_ = 1152, INTER_ = 4304, H_ = 16, HD_ = 72;
constexpr int M_ = B_ * S_;          // 11664 tokens
constexpr int INTER_PAD = 4320;      // fc1 output K-dim padded to multiple of 32
constexpr float LN_EPS_ = 1e-6f;

// LDS tile geometry: 128(M or N) x 32(K) bf16, row stride 48 elems = 96B (16B aligned)
constexpr int TK = 32;
constexpr int TSTRIDE = 48;
constexpr int TILE_ELEMS = 128 * TSTRIDE;

#if __has_builtin(__builtin_amdgcn_global_load_async_to_lds_b128)
#define HAVE_ASYNC_LDS 1
#else
#define HAVE_ASYNC_LDS 0
#endif

#if __has_builtin(__builtin_amdgcn_s_wait_asynccnt)
#define WAIT_ASYNC0() __builtin_amdgcn_s_wait_asynccnt(0)
#else
#define WAIT_ASYNC0() asm volatile("s_wait_asynccnt 0" ::: "memory")
#endif

#if __has_builtin(__builtin_amdgcn_tanhf)
#define TANHF(x) __builtin_amdgcn_tanhf(x)
#else
#define TANHF(x) tanhf(x)
#endif

DEV bf16_t f2bf(float x) {
    unsigned u = __builtin_bit_cast(unsigned, x) + 0x8000u;  // round-half-up, cheap
    unsigned short hs = (unsigned short)(u >> 16);
    return __builtin_bit_cast(bf16_t, hs);
}

// pack two fp32 -> two bf16 in one dword via v_perm_b32
DEV unsigned pkbf2(float a, float b) {
    unsigned ua = __builtin_bit_cast(unsigned, a) + 0x8000u;
    unsigned ub = __builtin_bit_cast(unsigned, b) + 0x8000u;
    return __builtin_amdgcn_perm(ub, ua, 0x07060302u);
}

// ---------------- bf16 pre-conversion (one pass per operand) ----------------
__global__ __launch_bounds__(256)
void cvt_bf16_kernel(const float* __restrict__ src, bf16_t* __restrict__ dst,
                     long src_ld, long dst_ld) {
    long row = blockIdx.y;
    long c0 = (long)blockIdx.x * 1024 + (long)threadIdx.x * 4;
    if (c0 >= dst_ld) return;
    uint2 pk;
    if (c0 + 3 < src_ld) {
        float4 f = *(const float4*)(src + row * src_ld + c0);
        pk.x = pkbf2(f.x, f.y);
        pk.y = pkbf2(f.z, f.w);
    } else {
        const float* p = src + row * src_ld;
        float t0 = (c0 + 0 < src_ld) ? p[c0 + 0] : 0.f;
        float t1 = (c0 + 1 < src_ld) ? p[c0 + 1] : 0.f;
        float t2 = (c0 + 2 < src_ld) ? p[c0 + 2] : 0.f;
        float t3 = (c0 + 3 < src_ld) ? p[c0 + 3] : 0.f;
        pk.x = pkbf2(t0, t1);
        pk.y = pkbf2(t2, t3);
    }
    *(uint2*)(dst + row * dst_ld + c0) = pk;
}

// ---------------- staging helpers ----------------

#if HAVE_ASYNC_LDS
// Async DMA global -> LDS (no VGPR data), tracked by ASYNCcnt.
DEV void async_cp_b128(const bf16_t* g, bf16_t* l) {
    __builtin_amdgcn_global_load_async_to_lds_b128(
        (__attribute__((address_space(1))) v4i*)g,
        (__attribute__((address_space(3))) v4i*)l,
        0, 0);
}

DEV void stage_tile_async(const bf16_t* __restrict__ src, long ld, long row0, long row_max,
                          long k0, bf16_t* lds, int tid) {
#pragma unroll
    for (int i = 0; i < 2; ++i) {
        int c = tid + i * 256;
        int r = c >> 2;
        int kc = (c & 3) * 8;
        long gr = row0 + r;
        long rc = (gr < row_max - 1) ? gr : (row_max - 1);  // branchless clamp
        async_cp_b128(src + rc * ld + k0 + kc, lds + r * TSTRIDE + kc);
    }
}
#endif

DEV void load_tile_regs(const bf16_t* __restrict__ src, long ld, long row0, long row_max,
                        long k0, int tid, uint4* regs) {
#pragma unroll
    for (int i = 0; i < 2; ++i) {
        int c = tid + i * 256;
        long r = row0 + (c >> 2);
        long rc = (r < row_max - 1) ? r : (row_max - 1);
        int kc = (c & 3) * 8;
        regs[i] = *(const uint4*)(src + rc * ld + k0 + kc);
    }
}

DEV void store_tile_lds(bf16_t* lds, int tid, const uint4* regs) {
#pragma unroll
    for (int i = 0; i < 2; ++i) {
        int c = tid + i * 256;
        int r = c >> 2, kc = (c & 3) * 8;
        *(uint4*)(lds + r * TSTRIDE + kc) = regs[i];
    }
}

// A fragment (16x32 bf16): lane<16 -> K {0..7,16..23}; lane>=16 -> K {8..15,24..31}
DEV v16bf load_fragA(const bf16_t* row_ptr, int lane) {
    int kb = (lane & 16) ? 8 : 0;
    union { v16bf v; uint4 u[2]; } f;
    f.u[0] = *(const uint4*)(row_ptr + kb);
    f.u[1] = *(const uint4*)(row_ptr + kb + 16);
    return f.v;
}

// B fragment (32x16 bf16), LDS tile N-major: lane = column; lanes<16 K {0..7,16..23}, lanes>=16 K {8..15,24..31}
DEV v16bf load_fragB(const bf16_t* col_ptr, int lane) {
    int kb = (lane & 16) ? 8 : 0;
    union { v16bf v; uint4 u[2]; } f;
    f.u[0] = *(const uint4*)(col_ptr + kb);
    f.u[1] = *(const uint4*)(col_ptr + kb + 16);
    return f.v;
}

// One k-step of WMMA work for this wave: 2 A-frags held live, B-frags streamed.
DEV void wmma_step(const bf16_t* Ac, const bf16_t* Bc, int wm, int wn, int lane, v8f acc[2][4]) {
    v16bf af0 = load_fragA(&Ac[(wm + 0  + (lane & 15)) * TSTRIDE], lane);
    v16bf af1 = load_fragA(&Ac[(wm + 16 + (lane & 15)) * TSTRIDE], lane);
#pragma unroll
    for (int ni = 0; ni < 4; ++ni) {
        v16bf bfv = load_fragB(&Bc[(wn + ni * 16 + (lane & 15)) * TSTRIDE], lane);
        acc[0][ni] = __builtin_amdgcn_wmma_f32_16x16x32_bf16(
            false, af0, false, bfv, (short)0, acc[0][ni], false, false);
        acc[1][ni] = __builtin_amdgcn_wmma_f32_16x16x32_bf16(
            false, af1, false, bfv, (short)0, acc[1][ni], false, false);
    }
}

// 128x128 block tile, 8 waves, each wave 32(M) x 64(N) = 2x4 fragments.
// Ping-pong buffers are addressed with compile-time constant offsets from smem so
// InferAddressSpaces keeps the fragment loads on the DS path (ds_load_b128).
DEV void gemm_core(const bf16_t* __restrict__ A, long ldA, long Mrows,
                   const bf16_t* __restrict__ Bm, long ldB, long Nrows,
                   int K, long m0, long n0, bf16_t* smem, int tid, v8f acc[2][4]) {
    int lane = tid & 31, wave = tid >> 5;
    int wm = (wave >> 1) * 32, wn = (wave & 1) * 64;
#if HAVE_ASYNC_LDS
    bf16_t* A0 = smem;
    bf16_t* B0 = smem + TILE_ELEMS;
    bf16_t* A1 = smem + 2 * TILE_ELEMS;
    bf16_t* B1 = smem + 3 * TILE_ELEMS;
    const int nk = K / TK;
    stage_tile_async(A, ldA, m0, Mrows, 0, A0, tid);
    stage_tile_async(Bm, ldB, n0, Nrows, 0, B0, tid);
    WAIT_ASYNC0();
    __syncthreads();
    for (int s = 0; s < nk; s += 2) {
        // even step: compute buf0, DMA next into buf1
        if (s + 1 < nk) {
            stage_tile_async(A, ldA, m0, Mrows, (long)(s + 1) * TK, A1, tid);
            stage_tile_async(Bm, ldB, n0, Nrows, (long)(s + 1) * TK, B1, tid);
        }
        wmma_step(A0, B0, wm, wn, lane, acc);
        WAIT_ASYNC0();
        __syncthreads();
        if (s + 1 >= nk) break;
        // odd step: compute buf1, DMA next into buf0
        if (s + 2 < nk) {
            stage_tile_async(A, ldA, m0, Mrows, (long)(s + 2) * TK, A0, tid);
            stage_tile_async(Bm, ldB, n0, Nrows, (long)(s + 2) * TK, B0, tid);
        }
        wmma_step(A1, B1, wm, wn, lane, acc);
        WAIT_ASYNC0();
        __syncthreads();
    }
#else
    // Fallback: register-prefetch pipeline, single LDS buffer.
    bf16_t* Alds = smem;
    bf16_t* Blds = smem + TILE_ELEMS;
    uint4 pa[2], pb[2];
    load_tile_regs(A, ldA, m0, Mrows, 0, tid, pa);
    load_tile_regs(Bm, ldB, n0, Nrows, 0, tid, pb);
    for (int k0 = 0; k0 < K; k0 += TK) {
        store_tile_lds(Alds, tid, pa);
        store_tile_lds(Blds, tid, pb);
        __syncthreads();
        if (k0 + TK < K) {
            load_tile_regs(A, ldA, m0, Mrows, k0 + TK, tid, pa);
            load_tile_regs(Bm, ldB, n0, Nrows, k0 + TK, tid, pb);
        }
        wmma_step(Alds, Blds, wm, wn, lane, acc);
        __syncthreads();
    }
#endif
}

#if HAVE_ASYNC_LDS
#define GEMM_SMEM_ELEMS (4 * TILE_ELEMS)
#else
#define GEMM_SMEM_ELEMS (2 * TILE_ELEMS)
#endif

// ---------------- GEMM kernels ----------------

__global__ __launch_bounds__(256)
void gemm_v_kernel(const bf16_t* __restrict__ hidb, const bf16_t* __restrict__ wvb,
                   const float* __restrict__ bv, float* __restrict__ vout) {
    __shared__ bf16_t smem[GEMM_SMEM_ELEMS];
    int tid = threadIdx.x, lane = tid & 31, wave = tid >> 5;
    int wm = (wave >> 1) * 32, wn = (wave & 1) * 64;
    long m0 = (long)blockIdx.y * 128, n0 = (long)blockIdx.x * 128;
    v8f acc[2][4] = {};
    gemm_core(hidb, D_, M_, wvb, D_, D_, D_, m0, n0, smem, tid, acc);
    int cl = lane & 15, rb = (lane & 16) ? 8 : 0;
#pragma unroll
    for (int ni = 0; ni < 4; ++ni) {
        long col = n0 + wn + ni * 16 + cl;
        float bias = bv[col];
#pragma unroll
        for (int mi = 0; mi < 2; ++mi) {
            long rbase = m0 + wm + mi * 16 + rb;
#pragma unroll
            for (int i = 0; i < 8; ++i) {
                long row = rbase + i;
                if (row < M_) vout[row * D_ + col] = acc[mi][ni][i] + bias;
            }
        }
    }
}

__global__ __launch_bounds__(256)
void gemm_fc1_kernel(const bf16_t* __restrict__ vn, const bf16_t* __restrict__ w1b,
                     const float* __restrict__ b1, bf16_t* __restrict__ hbuf) {
    __shared__ bf16_t smem[GEMM_SMEM_ELEMS];
    int tid = threadIdx.x, lane = tid & 31, wave = tid >> 5;
    int wm = (wave >> 1) * 32, wn = (wave & 1) * 64;
    long m0 = (long)blockIdx.y * 128, n0 = (long)blockIdx.x * 128;
    v8f acc[2][4] = {};
    gemm_core(vn, D_, M_, w1b, D_, INTER_, D_, m0, n0, smem, tid, acc);
    int cl = lane & 15, rb = (lane & 16) ? 8 : 0;
#pragma unroll
    for (int ni = 0; ni < 4; ++ni) {
        long col = n0 + wn + ni * 16 + cl;
        if (col >= INTER_PAD) continue;
        float bias = (col < INTER_) ? b1[col] : 0.f;
#pragma unroll
        for (int mi = 0; mi < 2; ++mi) {
            long rbase = m0 + wm + mi * 16 + rb;
#pragma unroll
            for (int i = 0; i < 8; ++i) {
                long row = rbase + i;
                if (row < M_) {
                    float x = acc[mi][ni][i] + bias;
                    float g = 0.5f * x * (1.f + TANHF(0.7978845608028654f * (x + 0.044715f * x * x * x)));
                    hbuf[row * INTER_PAD + col] = (col < INTER_) ? f2bf(g) : f2bf(0.f);
                }
            }
        }
    }
}

__global__ __launch_bounds__(256)
void gemm_fc2_kernel(const bf16_t* __restrict__ hbuf, const bf16_t* __restrict__ w2b,
                     const float* __restrict__ b2, float* __restrict__ out) {
    __shared__ bf16_t smem[GEMM_SMEM_ELEMS];
    int tid = threadIdx.x, lane = tid & 31, wave = tid >> 5;
    int wm = (wave >> 1) * 32, wn = (wave & 1) * 64;
    long m0 = (long)blockIdx.y * 128, n0 = (long)blockIdx.x * 128;
    v8f acc[2][4] = {};
    gemm_core(hbuf, INTER_PAD, M_, w2b, INTER_PAD, D_, INTER_PAD, m0, n0, smem, tid, acc);
    int cl = lane & 15, rb = (lane & 16) ? 8 : 0;
#pragma unroll
    for (int ni = 0; ni < 4; ++ni) {
        long col = n0 + wn + ni * 16 + cl;
        float bias = b2[col];
#pragma unroll
        for (int mi = 0; mi < 2; ++mi) {
            long rbase = m0 + wm + mi * 16 + rb;
#pragma unroll
            for (int i = 0; i < 8; ++i) {
                long row = rbase + i;
                if (row < M_) {
                    long idx = row * D_ + col;
                    out[idx] = acc[mi][ni][i] + bias + out[idx];  // residual: v + mlp
                }
            }
        }
    }
}

// ---------------- small kernels ----------------

__global__ __launch_bounds__(128)
void q_kernel(const float* __restrict__ probe, const float* __restrict__ Wq,
              const float* __restrict__ bq, float* __restrict__ q) {
    __shared__ float red[128];
    int d = blockIdx.x, tid = threadIdx.x;
    const float* wr = Wq + (long)d * D_;
    float s = 0.f;
    for (int j = tid; j < D_; j += 128) s += probe[j] * wr[j];
    red[tid] = s; __syncthreads();
    for (int o = 64; o > 0; o >>= 1) { if (tid < o) red[tid] += red[tid + o]; __syncthreads(); }
    if (tid == 0) q[d] = red[0] + bq[d];
}

__global__ __launch_bounds__(128)
void weff_kernel(const float* __restrict__ q, const float* __restrict__ Wk,
                 const float* __restrict__ bk, float* __restrict__ weff,
                 float* __restrict__ sbias) {
    __shared__ float qh[HD_];
    int h = blockIdx.y, tid = threadIdx.x;
    if (tid < HD_) qh[tid] = q[h * HD_ + tid] * 0.11785113019775793f;  // 1/sqrt(72)
    __syncthreads();
    int j = blockIdx.x * 128 + tid;
    float s = 0.f;
    for (int i = 0; i < HD_; ++i) s += qh[i] * Wk[(long)(h * HD_ + i) * D_ + j];
    weff[h * D_ + j] = s;
    if (blockIdx.x == 0 && tid == 0) {
        float sb = 0.f;
        for (int i = 0; i < HD_; ++i) sb += qh[i] * bk[h * HD_ + i];
        sbias[h] = sb;
    }
}

__global__ __launch_bounds__(256)
void scores_kernel(const float* __restrict__ hid, const float* __restrict__ weff,
                   const float* __restrict__ sbias, float* __restrict__ scores) {
    __shared__ float row[D_];
    long t = blockIdx.x;                 // token = b*S + s
    int tid = threadIdx.x;
    const float* src = hid + t * D_;
    for (int j = tid; j < D_; j += 256) row[j] = src[j];
    __syncthreads();
    int h = tid >> 4, l = tid & 15;
    float s = 0.f;
    for (int j = l; j < D_; j += 16) s += row[j] * weff[h * D_ + j];
#pragma unroll
    for (int o = 8; o > 0; o >>= 1) s += __shfl_xor(s, o, 16);
    if (l == 0) {
        long b = t / S_, ss = t % S_;
        scores[(b * H_ + h) * S_ + ss] = s + sbias[h];
    }
}

__global__ __launch_bounds__(256)
void softmax_kernel(const float* __restrict__ scores, float* __restrict__ attn) {
    __shared__ float e[S_];
    __shared__ float red[256];
    int bh = blockIdx.x, tid = threadIdx.x;
    const float* src = scores + (long)bh * S_;
    float m = -1e30f;
    for (int s = tid; s < S_; s += 256) m = fmaxf(m, src[s]);
    red[tid] = m; __syncthreads();
    for (int o = 128; o > 0; o >>= 1) { if (tid < o) red[tid] = fmaxf(red[tid], red[tid + o]); __syncthreads(); }
    m = red[0]; __syncthreads();
    float sum = 0.f;
    for (int s = tid; s < S_; s += 256) { float ex = __expf(src[s] - m); e[s] = ex; sum += ex; }
    red[tid] = sum; __syncthreads();
    for (int o = 128; o > 0; o >>= 1) { if (tid < o) red[tid] += red[tid + o]; __syncthreads(); }
    float inv = 1.f / red[0];
    float* dst = attn + (long)bh * S_;
    for (int s = tid; s < S_; s += 256) dst[s] = e[s] * inv;
}

__global__ __launch_bounds__(256)
void layernorm_kernel(const float* __restrict__ v, const float* __restrict__ w,
                      const float* __restrict__ bias, bf16_t* __restrict__ vn) {
    __shared__ float red[256];
    long t = blockIdx.x;
    const float* row = v + t * D_;
    int tid = threadIdx.x;
    float s = 0.f, s2 = 0.f;
    for (int j = tid; j < D_; j += 256) { float x = row[j]; s += x; s2 += x * x; }
    red[tid] = s; __syncthreads();
    for (int o = 128; o > 0; o >>= 1) { if (tid < o) red[tid] += red[tid + o]; __syncthreads(); }
    float mean = red[0] / D_; __syncthreads();
    red[tid] = s2; __syncthreads();
    for (int o = 128; o > 0; o >>= 1) { if (tid < o) red[tid] += red[tid + o]; __syncthreads(); }
    float var = red[0] / D_ - mean * mean;
    float rstd = rsqrtf(var + LN_EPS_);
    for (int j = tid; j < D_; j += 256) {
        float y = (row[j] - mean) * rstd * w[j] + bias[j];
        vn[t * D_ + j] = f2bf(y);
    }
}

// ---------------- launch ----------------

extern "C" void kernel_launch(void* const* d_in, const int* in_sizes, int n_in,
                              void* d_out, int out_size, void* d_ws, size_t ws_size,
                              hipStream_t stream) {
    (void)in_sizes; (void)n_in; (void)out_size; (void)ws_size;
    const float* hid   = (const float*)d_in[0];
    const float* probe = (const float*)d_in[1];
    const float* ipw   = (const float*)d_in[2];
    const float* ipb   = (const float*)d_in[3];
    const float* lnw   = (const float*)d_in[4];
    const float* lnb   = (const float*)d_in[5];
    const float* w1    = (const float*)d_in[6];
    const float* b1    = (const float*)d_in[7];
    const float* w2    = (const float*)d_in[8];
    const float* b2    = (const float*)d_in[9];

    const float* Wq = ipw;
    const float* Wk = ipw + (size_t)D_ * D_;
    const float* Wv = ipw + 2 * (size_t)D_ * D_;
    const float* bq = ipb;
    const float* bk = ipb + D_;
    const float* bv = ipb + 2 * D_;

    float* out_main = (float*)d_out;                   // (B,S,D)
    float* out_attn = out_main + (size_t)M_ * D_;      // (B,H,S)

    char* ws = (char*)d_ws;
    size_t off = 0;
    auto wsalloc = [&](size_t bytes) {
        void* p = ws + off;
        off = (off + bytes + 255) & ~(size_t)255;
        return p;
    };
    float*  qv     = (float*)wsalloc((size_t)D_ * sizeof(float));
    float*  sbias  = (float*)wsalloc((size_t)H_ * sizeof(float));
    float*  weff   = (float*)wsalloc((size_t)H_ * D_ * sizeof(float));
    float*  scores = (float*)wsalloc((size_t)B_ * H_ * S_ * sizeof(float));
    bf16_t* vn     = (bf16_t*)wsalloc((size_t)M_ * D_ * sizeof(bf16_t));
    bf16_t* hbuf   = (bf16_t*)wsalloc((size_t)M_ * INTER_PAD * sizeof(bf16_t));
    bf16_t* hidb   = (bf16_t*)wsalloc((size_t)M_ * D_ * sizeof(bf16_t));
    bf16_t* wvb    = (bf16_t*)wsalloc((size_t)D_ * D_ * sizeof(bf16_t));
    bf16_t* w1b    = (bf16_t*)wsalloc((size_t)INTER_ * D_ * sizeof(bf16_t));
    bf16_t* w2b    = (bf16_t*)wsalloc((size_t)D_ * INTER_PAD * sizeof(bf16_t));

    // one-time bf16 conversions (K of fc2 padded 4304 -> 4320 with zeros)
    cvt_bf16_kernel<<<dim3(2, M_),     256, 0, stream>>>(hid, hidb, D_, D_);
    cvt_bf16_kernel<<<dim3(2, D_),     256, 0, stream>>>(Wv, wvb, D_, D_);
    cvt_bf16_kernel<<<dim3(2, INTER_), 256, 0, stream>>>(w1, w1b, D_, D_);
    cvt_bf16_kernel<<<dim3(5, D_),     256, 0, stream>>>(w2, w2b, INTER_, INTER_PAD);

    // attention-weights path (k-projection folded into a 16-row effective weight)
    q_kernel<<<D_, 128, 0, stream>>>(probe, Wq, bq, qv);
    weff_kernel<<<dim3(D_ / 128, H_), 128, 0, stream>>>(qv, Wk, bk, weff, sbias);
    scores_kernel<<<M_, 256, 0, stream>>>(hid, weff, sbias, scores);
    softmax_kernel<<<B_ * H_, 256, 0, stream>>>(scores, out_attn);

    // main path: v -> LN -> fc1+gelu -> fc2 + residual
    gemm_v_kernel<<<dim3(D_ / 128, (M_ + 127) / 128), 256, 0, stream>>>(hidb, wvb, bv, out_main);
    layernorm_kernel<<<M_, 256, 0, stream>>>(out_main, lnw, lnb, vn);
    gemm_fc1_kernel<<<dim3((INTER_ + 127) / 128, (M_ + 127) / 128), 256, 0, stream>>>(vn, w1b, b1, hbuf);
    gemm_fc2_kernel<<<dim3(D_ / 128, (M_ + 127) / 128), 256, 0, stream>>>(hbuf, w2b, b2, out_main);
}
